// GSModel_72284299592413
// MI455X (gfx1250) — compile-verified, matrix operand order checked by reading, not run on hardware
//
#include <hip/hip_runtime.h>

// ---------------------------------------------------------------------------
// GraphSAGE 2-layer forward for MI455X (gfx1250, wave32, WMMA bf16)
//   h0=feats[ids], m1=mean25(feats[ids1]), h1=feats[ids1], m2=mean10(feats[ids2])
//   g1 = relu([h1@Wx1 | m2@Wn1])   (12800 x 256)
//   g0 = relu([h0@Wx1 | m1@Wn1])   (512 x 256)
//   o  = [g0@Wx2 | mean25(g1)@Wn2] -> row-normalize -> @fcW + fcb  (512 x 41)
// All GEMMs via v_wmma_f32_16x16x32_bf16 (f32 accumulate).
// ---------------------------------------------------------------------------

typedef __bf16 bf16_t;
typedef __attribute__((ext_vector_type(16))) __bf16 v16bf;
typedef __attribute__((ext_vector_type(8)))  float  v8f;

#define IN_DIM   602
#define KPAD1    608      // 19 * 32
#define HDIM     128
#define HH       256      // 2*H
#define BATCH    512
#define FAN1     25
#define FAN2     10
#define NCLASS   41
#define R1       (BATCH * FAN1)   // 12800

__device__ __forceinline__ bf16_t f2bf(float f) {
  unsigned u = __builtin_bit_cast(unsigned, f);
  u += 0x7FFFu + ((u >> 16) & 1u);              // round-to-nearest-even
  unsigned short h = (unsigned short)(u >> 16);
  return __builtin_bit_cast(bf16_t, h);
}
__device__ __forceinline__ float bf2f(bf16_t b) {
  unsigned short h = __builtin_bit_cast(unsigned short, b);
  unsigned u = (unsigned)h << 16;
  return __builtin_bit_cast(float, u);
}

// Per-lane K offset for the 16-bit 16x32 A/B fragment layout (ISA 7.12.2):
// lanes 0-15 hold K {0..7,16..23}, lanes 16-31 hold K {8..15,24..31},
// two consecutive K per 32-bit VGPR.
__device__ __forceinline__ int frag_k(int v, int hi) {
  return ((v >> 2) << 4) + (hi << 3) + ((v & 3) << 1);
}

// ---------------- weight transpose + pad + bf16 convert --------------------
// W[K][N] f32  ->  Wt[N][Kpad] bf16 (zero-padded K)
__global__ void wtrans_kernel(const float* __restrict__ W, bf16_t* __restrict__ Wt,
                              int K, int N, int Kpad) {
  int idx = blockIdx.x * blockDim.x + threadIdx.x;
  if (idx >= N * Kpad) return;
  int n = idx / Kpad, k = idx - n * Kpad;
  float v = (k < K) ? W[(size_t)k * N + n] : 0.0f;
  Wt[idx] = f2bf(v);
}

// ---------------- gather one feats row -> bf16 padded row ------------------
__global__ __launch_bounds__(256) void gather_kernel(
    const float* __restrict__ feats, const int* __restrict__ ids,
    bf16_t* __restrict__ out) {
  int i = blockIdx.x;
  size_t src = (size_t)ids[i] * IN_DIM;
  for (int c = threadIdx.x; c < KPAD1; c += 256) {
    float v = (c < IN_DIM) ? feats[src + c] : 0.0f;
    out[(size_t)i * KPAD1 + c] = f2bf(v);
  }
}

// ---------------- fused fanout-mean of gathered feats rows -----------------
__global__ __launch_bounds__(256) void mean_feats_kernel(
    const float* __restrict__ feats, const int* __restrict__ ids,
    bf16_t* __restrict__ out, int fan) {
  __shared__ int sid[32];
  int i = blockIdx.x;
  if (threadIdx.x < fan) sid[threadIdx.x] = ids[i * fan + threadIdx.x];
  __syncthreads();
  float inv = 1.0f / (float)fan;
  for (int c = threadIdx.x; c < KPAD1; c += 256) {
    float s = 0.0f;
    if (c < IN_DIM) {
      for (int j = 0; j < fan; ++j) s += feats[(size_t)sid[j] * IN_DIM + c];
    }
    out[(size_t)i * KPAD1 + c] = f2bf(s * inv);
  }
}

// ---------------- layer-1 style GEMM: G[:,half] = relu(A @ W) --------------
// A: [nrows][kpad] bf16 (h or mean), Wt: [128][kpad] bf16 (transposed weight)
// 256 threads = 8 wave32; block computes a 16-row x 128-col half of G[.,256].
// blockIdx.y: 0 -> (A0,W0t) cols 0..127 ; 1 -> (A1,W1t) cols 128..255
__global__ __launch_bounds__(256) void gemm_tile_kernel(
    const bf16_t* __restrict__ A0, const bf16_t* __restrict__ A1,
    const bf16_t* __restrict__ W0t, const bf16_t* __restrict__ W1t,
    bf16_t* __restrict__ G, int kpad, int kTiles, int doRelu) {
  __shared__ bf16_t sA[16 * KPAD1];
  const bf16_t* A  = (blockIdx.y == 0) ? A0  : A1;
  const bf16_t* Wt = (blockIdx.y == 0) ? W0t : W1t;
  const int colOff = (blockIdx.y == 0) ? 0 : HDIM;
  const int row0 = blockIdx.x * 16;
  const int tid  = threadIdx.x;

  // stage the 16-row A tile into LDS (coalesced)
  const int total = 16 * kpad;
  for (int idx = tid; idx < total; idx += 256) {
    int r = idx / kpad, k = idx - r * kpad;
    sA[r * kpad + k] = A[(size_t)(row0 + r) * kpad + k];
  }
  __syncthreads();

  const int lane = tid & 31;
  const int wave = tid >> 5;              // wave owns 16 output cols
  const int n    = lane & 15;
  const int hi   = lane >> 4;
  const bf16_t* arow = sA + (size_t)n * kpad;                 // A row M = lane&15
  const bf16_t* brow = Wt + (size_t)(wave * 16 + n) * kpad;   // B col N = lane&15

  v8f acc = {};
  for (int kt = 0; kt < kTiles; ++kt) {
    int kb = kt * 32;
    v16bf a, b;
#pragma unroll
    for (int v = 0; v < 8; ++v) {
      int k = kb + frag_k(v, hi);
      a[2 * v]     = arow[k];
      a[2 * v + 1] = arow[k + 1];
      b[2 * v]     = brow[k];
      b[2 * v + 1] = brow[k + 1];
    }
    acc = __builtin_amdgcn_wmma_f32_16x16x32_bf16(
        false, a, false, b, (short)0, acc, false, false);
  }

  // C/D layout: lane n = lane&15, rows (lane>>4)*8 + r in VGPR r
  const int mb = hi * 8;
  const int cb = colOff + wave * 16 + n;
#pragma unroll
  for (int r = 0; r < 8; ++r) {
    float x = acc[r];
    if (doRelu) x = fmaxf(x, 0.0f);
    G[(size_t)(row0 + mb + r) * HH + cb] = f2bf(x);
  }
}

// ---------------- mean of g1 over fanout 25 --------------------------------
__global__ void mean_g1_kernel(const bf16_t* __restrict__ G1,
                               bf16_t* __restrict__ MG1) {
  int idx = blockIdx.x * blockDim.x + threadIdx.x;
  if (idx >= BATCH * HH) return;
  int b = idx >> 8, c = idx & 255;
  float s = 0.0f;
  for (int j = 0; j < FAN1; ++j)
    s += bf2f(G1[(size_t)(b * FAN1 + j) * HH + c]);
  MG1[idx] = f2bf(s * (1.0f / FAN1));
}

// ---------------- layer 2 + row-normalize + fc, fully fused ----------------
// One block per 16 output rows. o[16][256] built with WMMA into LDS, then
// norms + (o/||o||) @ fcW + fcb  ->  out[16][41].
__global__ __launch_bounds__(256) void layer2_kernel(
    const bf16_t* __restrict__ G0, const bf16_t* __restrict__ MG1,
    const bf16_t* __restrict__ Wx2t, const bf16_t* __restrict__ Wn2t,
    const float* __restrict__ fcW, const float* __restrict__ fcB,
    float* __restrict__ out) {
  __shared__ bf16_t sA[2][16 * HH];
  __shared__ float  sO[16][HH + 8];
  __shared__ float  sNorm[16];
  const int row0 = blockIdx.x * 16;
  const int tid  = threadIdx.x;

  for (int idx = tid; idx < 16 * HH; idx += 256) {
    int r = idx >> 8, c = idx & 255;
    sA[0][idx] = G0 [(size_t)(row0 + r) * HH + c];
    sA[1][idx] = MG1[(size_t)(row0 + r) * HH + c];
  }
  __syncthreads();

  const int lane = tid & 31, wave = tid >> 5;
  const int n = lane & 15, hi = lane >> 4;
#pragma unroll
  for (int i = 0; i < 2; ++i) {
    int t    = wave * 2 + i;          // 16 col-tiles of 16
    int half = t >> 3;                // 0: g0@Wx2 | 1: mg1@Wn2
    int bcol = (t & 7) * 16;
    const bf16_t* arow = &sA[half][(size_t)n * HH];
    const bf16_t* brow = (half ? Wn2t : Wx2t) + (size_t)(bcol + n) * HH;
    v8f acc = {};
#pragma unroll
    for (int kt = 0; kt < 8; ++kt) {
      int kb = kt * 32;
      v16bf a, b;
#pragma unroll
      for (int v = 0; v < 8; ++v) {
        int k = kb + frag_k(v, hi);
        a[2 * v]     = arow[k];
        a[2 * v + 1] = arow[k + 1];
        b[2 * v]     = brow[k];
        b[2 * v + 1] = brow[k + 1];
      }
      acc = __builtin_amdgcn_wmma_f32_16x16x32_bf16(
          false, a, false, b, (short)0, acc, false, false);
    }
    const int mb = hi * 8, cb = t * 16 + n;
#pragma unroll
    for (int r = 0; r < 8; ++r) sO[mb + r][cb] = acc[r];
  }
  __syncthreads();

  if (tid < 16) {
    float s = 0.0f;
    for (int c = 0; c < HH; ++c) { float x = sO[tid][c]; s += x * x; }
    sNorm[tid] = fmaxf(sqrtf(s), 1e-12f);
  }
  __syncthreads();

  for (int idx = tid; idx < 16 * NCLASS; idx += 256) {
    int r = idx / NCLASS, c = idx - r * NCLASS;
    float inv = 1.0f / sNorm[r];
    float s = fcB[c];
    for (int k = 0; k < HH; ++k)
      s += sO[r][k] * inv * fcW[k * NCLASS + c];
    out[(size_t)(row0 + r) * NCLASS + c] = s;
  }
}

// ---------------------------------------------------------------------------
extern "C" void kernel_launch(void* const* d_in, const int* in_sizes, int n_in,
                              void* d_out, int out_size, void* d_ws, size_t ws_size,
                              hipStream_t stream) {
  (void)in_sizes; (void)n_in; (void)out_size; (void)ws_size;
  const int*   ids   = (const int*)  d_in[0];
  const int*   ids1  = (const int*)  d_in[1];
  const int*   ids2  = (const int*)  d_in[2];
  const float* feats = (const float*)d_in[3];
  const float* Wx1   = (const float*)d_in[4];
  const float* Wn1   = (const float*)d_in[5];
  const float* Wx2   = (const float*)d_in[6];
  const float* Wn2   = (const float*)d_in[7];
  const float* fcW   = (const float*)d_in[8];
  const float* fcB   = (const float*)d_in[9];
  float* out = (float*)d_out;

  char* ws = (char*)d_ws;
  size_t off = 0;
  auto alloc = [&](size_t bytes) {
    char* p = ws + off;
    off = (off + bytes + 255) & ~(size_t)255;
    return p;
  };
  bf16_t* h1   = (bf16_t*)alloc((size_t)R1    * KPAD1 * 2);  // 15.6 MB
  bf16_t* m2   = (bf16_t*)alloc((size_t)R1    * KPAD1 * 2);  // 15.6 MB
  bf16_t* h0   = (bf16_t*)alloc((size_t)BATCH * KPAD1 * 2);
  bf16_t* m1   = (bf16_t*)alloc((size_t)BATCH * KPAD1 * 2);
  bf16_t* Wx1t = (bf16_t*)alloc((size_t)HDIM  * KPAD1 * 2);
  bf16_t* Wn1t = (bf16_t*)alloc((size_t)HDIM  * KPAD1 * 2);
  bf16_t* Wx2t = (bf16_t*)alloc((size_t)HDIM  * HH    * 2);
  bf16_t* Wn2t = (bf16_t*)alloc((size_t)HDIM  * HH    * 2);
  bf16_t* g1   = (bf16_t*)alloc((size_t)R1    * HH    * 2);  // 6.6 MB
  bf16_t* g0   = (bf16_t*)alloc((size_t)BATCH * HH    * 2);
  bf16_t* mg1  = (bf16_t*)alloc((size_t)BATCH * HH    * 2);

  // 1) weight transpose/pad/convert (tiny)
  int w1n = HDIM * KPAD1;
  wtrans_kernel<<<(w1n + 255) / 256, 256, 0, stream>>>(Wx1, Wx1t, IN_DIM, HDIM, KPAD1);
  wtrans_kernel<<<(w1n + 255) / 256, 256, 0, stream>>>(Wn1, Wn1t, IN_DIM, HDIM, KPAD1);
  int w2n = HDIM * HH;
  wtrans_kernel<<<(w2n + 255) / 256, 256, 0, stream>>>(Wx2, Wx2t, HH, HDIM, HH);
  wtrans_kernel<<<(w2n + 255) / 256, 256, 0, stream>>>(Wn2, Wn2t, HH, HDIM, HH);

  // 2) gathers + fused fanout means (the 308 MB h2 gather collapses into m2)
  gather_kernel    <<<BATCH, 256, 0, stream>>>(feats, ids,  h0);
  gather_kernel    <<<R1,    256, 0, stream>>>(feats, ids1, h1);
  mean_feats_kernel<<<BATCH, 256, 0, stream>>>(feats, ids1, m1, FAN1);
  mean_feats_kernel<<<R1,    256, 0, stream>>>(feats, ids2, m2, FAN2);

  // 3) layer-1 GEMMs (bf16 WMMA): g1 = relu([h1@Wx1 | m2@Wn1]), g0 likewise
  gemm_tile_kernel<<<dim3(R1 / 16, 2),    256, 0, stream>>>(
      h1, m2, Wx1t, Wn1t, g1, KPAD1, KPAD1 / 32, 1);
  gemm_tile_kernel<<<dim3(BATCH / 16, 2), 256, 0, stream>>>(
      h0, m1, Wx1t, Wn1t, g0, KPAD1, KPAD1 / 32, 1);

  // 4) neighbor mean of g1
  mean_g1_kernel<<<(BATCH * HH + 255) / 256, 256, 0, stream>>>(g1, mg1);

  // 5) layer 2 + normalize + fc, fused
  layer2_kernel<<<BATCH / 16, 256, 0, stream>>>(g0, mg1, Wx2t, Wn2t, fcW, fcB, out);
}
